// AudioEncoder_17239998726195
// MI455X (gfx1250) — compile-verified, matrix operand order checked by reading, not run on hardware
//
#include <hip/hip_runtime.h>
#include <math.h>

// ---------------------------------------------------------------------------
// Types for CDNA5 WMMA (wave32): v_wmma_f32_16x16x32_bf16
// ---------------------------------------------------------------------------
typedef __attribute__((ext_vector_type(16))) __bf16 v16bf;
typedef __attribute__((ext_vector_type(8)))  float  v8f;

union FragBF {
    uint4 q[2];   // 2 x 16B LDS loads
    v16bf v;      // 16 bf16 = 32B fragment
};

// fast transcendentals: single v_exp_f32 / v_log_f32 / v_rcp_f32
__device__ __forceinline__ float siluf(float x) {
    return __fdividef(x, 1.0f + __expf(-x));
}
// branchless, overflow-free softplus
__device__ __forceinline__ float softplusf(float x) {
    return fmaxf(x, 0.0f) + __logf(1.0f + __expf(-fabsf(x)));
}

// ---------------------------------------------------------------------------
// WMMA GEMM with pre-converted bf16 operands and double-buffered async
// global->LDS staging (GLOBAL_LOAD_ASYNC_TO_LDS_B128 / ASYNCcnt):
//   out[m][n] = act( Cin[m][n] + bias[n] + sum_k A[m][k]*W[n][k] )
//   A: bf16 row-major (lda elems), W: bf16 row-major [N][K] (ldw elems)
//   Cout (fp32) and/or Coutb (bf16) written if non-null.
// PRE-CONDITIONS: grid tiles M x N exactly (multiples of 64), K mult of 64,
//   lda/ldw multiples of 8 (16B-aligned rows).
// Block: 256 threads (8 wave32s), tile 64x64, K-step 64.
// ---------------------------------------------------------------------------
#define KSTEP   64
#define LDS_LD  72   // shorts per LDS row: 64 data + 8 pad (144B, 16B-aligned)

__global__ __launch_bounds__(256) void gemm_wmma_bf16(
    const __bf16* __restrict__ A, int lda,
    const __bf16* __restrict__ W, int ldw,
    const float* __restrict__ Cin,
    const float* __restrict__ bias,
    float* __restrict__ Cout, __bf16* __restrict__ Coutb, int ldc,
    int K, int act)
{
    __shared__ __align__(16) unsigned short As[2][64 * LDS_LD];
    __shared__ __align__(16) unsigned short Ws[2][64 * LDS_LD];

    const int tid  = threadIdx.x;
    const int lane = tid & 31;
    const int wave = tid >> 5;

    const int m_blk = blockIdx.y * 64;
    const int n_blk = blockIdx.x * 64;
    const int mw = (wave >> 1) * 16;          // 0,16,32,48
    const int nw = (wave & 1) * 32;           // 0,32

    const int half = lane >> 4;               // 0/1 (k-group select)
    const int l16  = lane & 15;
    const int lo   = half * 8;                // k sub-offset in LDS row

    const int out_n0 = n_blk + nw + l16;
    const int out_n1 = out_n0 + 16;

    // ---- accumulator init: residual + bias (uniform null-checks only)
    v8f acc0 = {}, acc1 = {};
    {
        const float b0 = bias ? bias[out_n0] : 0.0f;
        const float b1 = bias ? bias[out_n1] : 0.0f;
#pragma unroll
        for (int r = 0; r < 8; ++r) {
            const int out_m = m_blk + mw + r + 8 * half;
            float c0 = b0, c1 = b1;
            if (Cin) {
                c0 += Cin[(size_t)out_m * ldc + out_n0];
                c1 += Cin[(size_t)out_m * ldc + out_n1];
            }
            acc0[r] = c0; acc1[r] = c1;
        }
    }

    // staging map: thread t copies 16 consecutive bf16 (32B) of one row
    const int srow = tid >> 2;                // 0..63
    const int scol = (tid & 3) * 16;          // 0,16,32,48
    const int mg = m_blk + srow;
    const int ng = n_blk + srow;

    // LDS byte addresses (low 32 bits of generic pointer = LDS offset)
    const unsigned ldsA0 = (unsigned)(uintptr_t)&As[0][srow * LDS_LD + scol];
    const unsigned ldsA1 = (unsigned)(uintptr_t)&As[1][srow * LDS_LD + scol];
    const unsigned ldsW0 = (unsigned)(uintptr_t)&Ws[0][srow * LDS_LD + scol];
    const unsigned ldsW1 = (unsigned)(uintptr_t)&Ws[1][srow * LDS_LD + scol];

    const __bf16* arow_g = &A[(size_t)mg * lda + scol];
    const __bf16* wrow_g = &W[(size_t)ng * ldw + scol];

    // issue 4 async b128 copies (32B of A + 32B of W) into buffer `b` for k0
    // ISA: INST_OFFSET is applied to BOTH the LDS and global addresses.
#define ISSUE_ASYNC(ldsA, ldsW, k0)                                          \
    {                                                                        \
        unsigned long long ga = (unsigned long long)(uintptr_t)(arow_g + (k0)); \
        unsigned long long gw = (unsigned long long)(uintptr_t)(wrow_g + (k0)); \
        asm volatile(                                                        \
            "global_load_async_to_lds_b128 %0, %2, off\n\t"                  \
            "global_load_async_to_lds_b128 %0, %2, off offset:16\n\t"        \
            "global_load_async_to_lds_b128 %1, %3, off\n\t"                  \
            "global_load_async_to_lds_b128 %1, %3, off offset:16"            \
            :: "v"(ldsA), "v"(ldsW), "v"(ga), "v"(gw) : "memory");           \
    }

    ISSUE_ASYNC(ldsA0, ldsW0, 0)

    int buf = 0;
    for (int k0 = 0; k0 < K; k0 += KSTEP) {
        asm volatile("s_wait_asynccnt 0" ::: "memory");   // our copies landed
        __syncthreads();                                   // everyone's landed

        // prefetch next K-tile into the other buffer while we compute
        if (k0 + KSTEP < K) {
            if (buf == 0) { ISSUE_ASYNC(ldsA1, ldsW1, k0 + KSTEP) }
            else          { ISSUE_ASYNC(ldsA0, ldsW0, k0 + KSTEP) }
        }

        // ---- two K=32 panels, each: 1 A-frag, 2 B-frags, 2 WMMA
        const unsigned short* arow  = &As[buf][(mw + l16) * LDS_LD];
        const unsigned short* brow0 = &Ws[buf][(nw + l16) * LDS_LD];
        const unsigned short* brow1 = &Ws[buf][(nw + 16 + l16) * LDS_LD];
#pragma unroll
        for (int p = 0; p < 2; ++p) {
            const int pb = p * 32;
            FragBF fa, fb0, fb1;
            fa.q[0]  = *(const uint4*)&arow[pb + lo];
            fa.q[1]  = *(const uint4*)&arow[pb + 16 + lo];
            fb0.q[0] = *(const uint4*)&brow0[pb + lo];
            fb0.q[1] = *(const uint4*)&brow0[pb + 16 + lo];
            fb1.q[0] = *(const uint4*)&brow1[pb + lo];
            fb1.q[1] = *(const uint4*)&brow1[pb + 16 + lo];

            acc0 = __builtin_amdgcn_wmma_f32_16x16x32_bf16(
                false, fa.v, false, fb0.v, (short)0, acc0, false, false);
            acc1 = __builtin_amdgcn_wmma_f32_16x16x32_bf16(
                false, fa.v, false, fb1.v, (short)0, acc1, false, false);
        }

        __syncthreads();     // all waves done reading `buf` before it's refilled
        buf ^= 1;
    }
#undef ISSUE_ASYNC

    // ---- epilogue (unguarded; grid tiles output exactly)
#pragma unroll
    for (int r = 0; r < 8; ++r) {
        const int out_m = m_blk + mw + r + 8 * half;
        float v0 = acc0[r], v1 = acc1[r];
        if (act == 1) {
            v0 = softplusf(v0);
            v1 = softplusf(v1);
        }
        if (Cout) {
            Cout[(size_t)out_m * ldc + out_n0] = v0;
            Cout[(size_t)out_m * ldc + out_n1] = v1;
        }
        if (Coutb) {
            Coutb[(size_t)out_m * ldc + out_n0] = (__bf16)v0;
            Coutb[(size_t)out_m * ldc + out_n1] = (__bf16)v1;
        }
    }
}

// ---------------------------------------------------------------------------
// Zero-pad + convert fp32 -> bf16: dst[r][c] = (r<srows && c<scols)?src:0
// ---------------------------------------------------------------------------
__global__ __launch_bounds__(256) void padcvt_kernel(
    const float* __restrict__ src, int srows, int scols,
    __bf16* __restrict__ dst, int dcols, int total)
{
    int idx = blockIdx.x * blockDim.x + threadIdx.x;
    if (idx >= total) return;
    const int r = idx / dcols;
    const int c = idx - r * dcols;
    const float v = (r < srows && c < scols) ? src[(size_t)r * scols + c] : 0.0f;
    dst[idx] = (__bf16)v;
}

// ---------------------------------------------------------------------------
// Causal depthwise conv1d (K=4) + bias + SiLU.  xz rows have 1024 floats;
// xc part is cols [0,512).  Outputs: xc fp32 (scan/gate) + xcb bf16 (GEMM).
// ---------------------------------------------------------------------------
__global__ __launch_bounds__(256) void conv_silu_kernel(
    const float* __restrict__ xz, const float* __restrict__ cw,
    const float* __restrict__ cb, float* __restrict__ xc,
    __bf16* __restrict__ xcb, int total)
{
    int idx = blockIdx.x * blockDim.x + threadIdx.x;
    if (idx >= total) return;
    const int d = idx & 511;
    const int m = idx >> 9;
    const int l = m & 1023;

    const float w0 = cw[d * 4 + 0], w1 = cw[d * 4 + 1];
    const float w2 = cw[d * 4 + 2], w3 = cw[d * 4 + 3];

    const float* base = xz + (size_t)m * 1024 + d;
    float acc = cb[d] + w3 * base[0];
    if (l >= 1) acc += w2 * base[-1 * 1024];
    if (l >= 2) acc += w1 * base[-2 * 1024];
    if (l >= 3) acc += w0 * base[-3 * 1024];
    const float r = siluf(acc);
    xc[idx]  = r;
    xcb[idx] = (__bf16)r;
}

// ---------------------------------------------------------------------------
// Selective scan: one thread per (b, d); h[16] in registers; 1024 steps.
// dbl rows (stride 64): [0..15]=dt-rank, [16..31]=B_t, [32..47]=C_t, [48..63]=0
// exp via v_exp_f32 TRANS op (co-executes with VALU) -- serial-latency critical.
// ---------------------------------------------------------------------------
__global__ __launch_bounds__(512) void scan_kernel(
    const float* __restrict__ dt, const float* __restrict__ xc,
    const float* __restrict__ dbl, const float* __restrict__ A_log,
    float* __restrict__ ys)
{
    const int b = blockIdx.x;
    const int d = threadIdx.x;

    float Arow[16];
#pragma unroll
    for (int s = 0; s < 16; ++s) Arow[s] = -__expf(A_log[d * 16 + s]);

    float h[16];
#pragma unroll
    for (int s = 0; s < 16; ++s) h[s] = 0.0f;

    int m = b * 1024;
    for (int l = 0; l < 1024; ++l, ++m) {
        const float dtv = dt[(size_t)m * 512 + d];
        const float xv  = xc[(size_t)m * 512 + d];
        const float dx  = dtv * xv;
        const float* bc = dbl + (size_t)m * 64;   // wave-uniform -> scalar loads
        if (l + 16 < 1024) {
            __builtin_prefetch(bc + 16 * 64, 0, 3);        // global_prefetch_b8
            __builtin_prefetch(&dt[(size_t)(m + 16) * 512 + d], 0, 3);
        }
        float y = 0.0f;
#pragma unroll
        for (int s = 0; s < 16; ++s) {
            const float dA = __expf(dtv * Arow[s]);        // v_exp_f32
            h[s] = h[s] * dA + dx * bc[16 + s];
            y += h[s] * bc[32 + s];
        }
        ys[(size_t)m * 512 + d] = y;
    }
}

// ---------------------------------------------------------------------------
// Gating: g = (ys + xc * D_vec[d]) * silu(z),  z = xz[m][512 + d]; bf16 out
// ---------------------------------------------------------------------------
__global__ __launch_bounds__(256) void gate_kernel(
    const float* __restrict__ ys, const float* __restrict__ xc,
    const float* __restrict__ xz, const float* __restrict__ D_vec,
    __bf16* __restrict__ g, int total)
{
    int idx = blockIdx.x * blockDim.x + threadIdx.x;
    if (idx >= total) return;
    const int d = idx & 511;
    const int m = idx >> 9;
    const float z = xz[(size_t)m * 1024 + 512 + d];
    g[idx] = (__bf16)((ys[idx] + xc[idx] * D_vec[d]) * siluf(z));
}

// ---------------------------------------------------------------------------
// LayerNorm over 256 features: one wave32 per row, 8 rows per block.
// Writes fp32 (out) and/or bf16 (outb) if non-null.
// ---------------------------------------------------------------------------
__global__ __launch_bounds__(256) void layernorm_kernel(
    const float* __restrict__ x, const float* __restrict__ g,
    const float* __restrict__ b, float* __restrict__ out,
    __bf16* __restrict__ outb, int nrows)
{
    const int wave = threadIdx.x >> 5;
    const int lane = threadIdx.x & 31;
    const int row  = blockIdx.x * 8 + wave;
    if (row >= nrows) return;

    const float* xr = x + (size_t)row * 256;
    float v[8];
    float s = 0.0f;
#pragma unroll
    for (int j = 0; j < 8; ++j) { v[j] = xr[lane + j * 32]; s += v[j]; }
#pragma unroll
    for (int off = 16; off > 0; off >>= 1) s += __shfl_xor(s, off, 32);
    const float mu = s * (1.0f / 256.0f);

    float vs = 0.0f;
#pragma unroll
    for (int j = 0; j < 8; ++j) { const float dlt = v[j] - mu; vs += dlt * dlt; }
#pragma unroll
    for (int off = 16; off > 0; off >>= 1) vs += __shfl_xor(vs, off, 32);
    const float inv = rsqrtf(vs * (1.0f / 256.0f) + 1e-5f);

#pragma unroll
    for (int j = 0; j < 8; ++j) {
        const int c = lane + j * 32;
        const float r = (v[j] - mu) * inv * g[c] + b[c];
        if (out)  out [(size_t)row * 256 + c] = r;
        if (outb) outb[(size_t)row * 256 + c] = (__bf16)r;
    }
}

// ---------------------------------------------------------------------------
// Host-side orchestration
// ---------------------------------------------------------------------------
static inline void launch_gemm(const __bf16* A, int lda, const __bf16* W, int ldw,
                               const float* Cin, const float* bias,
                               float* Cout, __bf16* Coutb, int ldc,
                               int M, int N, int K, int act, hipStream_t stream)
{
    dim3 grid(N / 64, M / 64);
    gemm_wmma_bf16<<<grid, 256, 0, stream>>>(A, lda, W, ldw, Cin, bias,
                                             Cout, Coutb, ldc, K, act);
}

static inline void launch_cvt(const float* src, int srows, int scols,
                              __bf16* dst, int drows, int dcols, hipStream_t stream)
{
    const int total = drows * dcols;
    padcvt_kernel<<<(total + 255) / 256, 256, 0, stream>>>(src, srows, scols,
                                                           dst, dcols, total);
}

extern "C" void kernel_launch(void* const* d_in, const int* in_sizes, int n_in,
                              void* d_out, int out_size, void* d_ws, size_t ws_size,
                              hipStream_t stream)
{
    (void)in_sizes; (void)n_in; (void)out_size; (void)ws_size;

    const float* wav     = (const float*)d_in[0];   // [8,1024,768]
    const float* lib     = (const float*)d_in[1];   // [8,1024,93]
    const float* w2v_W   = (const float*)d_in[2];   // [256,768]
    const float* w2v_b   = (const float*)d_in[3];
    const float* lib_W   = (const float*)d_in[4];   // [256,93]
    const float* lib_b   = (const float*)d_in[5];
    const float* fuse_W  = (const float*)d_in[6];   // [256,512]
    const float* fuse_b  = (const float*)d_in[7];
    const float* proj_W  = (const float*)d_in[8];   // [256,256]
    const float* proj_b  = (const float*)d_in[9];
    const float* ln_g    = (const float*)d_in[10];  // [4,256]
    const float* ln_b    = (const float*)d_in[11];
    const float* in_W    = (const float*)d_in[12];  // [4,1024,256]
    const float* conv_W  = (const float*)d_in[13];  // [4,512,4]
    const float* conv_b  = (const float*)d_in[14];  // [4,512]
    const float* xproj_W = (const float*)d_in[15];  // [4,48,512]
    const float* dt_W    = (const float*)d_in[16];  // [4,512,16]
    const float* dt_b    = (const float*)d_in[17];  // [4,512]
    const float* A_log   = (const float*)d_in[18];  // [4,512,16]
    const float* D_vec   = (const float*)d_in[19];  // [4,512]
    const float* out_W   = (const float*)d_in[20];  // [4,256,512]
    const float* fnorm_g = (const float*)d_in[21];
    const float* fnorm_b = (const float*)d_in[22];

    const int M = 8192;                 // B * L
    float* ws = (float*)d_ws;
    const size_t MEG = 1u << 20;        // 1M floats

    // ---- fp32 scratch ----
    float* h    = ws;                   // 2M   : residual stream [8192 x 256]
    float* xz   = ws + 2  * MEG;        // 8M   : [8192 x 1024]
    float* xc   = ws + 10 * MEG;        // 4M   : [8192 x 512]
    float* dtb  = ws + 14 * MEG;        // 4M   : dt
    float* ysb  = ws + 18 * MEG;        // 4M   : scan out
    float* fbuf = ws + 22 * MEG;        // 2M   : fuse accum [8192 x 256]
    float* dbl  = ws + 24 * MEG;        // 0.5M : [8192 x 64]

    // ---- bf16 scratch (offsets in float units; 1 float = 2 bf16) ----
    // window 24.5M..27.75M holds wavb early; later xcb/dblb/lnbb (disjoint in time)
    __bf16* wavb  = (__bf16*)(ws + 24 * MEG + MEG / 2);   // 8192x768  (3.07M fl)
    __bf16* xcb   = (__bf16*)(ws + 24 * MEG + MEG / 2);   // 8192x512  (2M fl)
    __bf16* dblb  = (__bf16*)(ws + 26 * MEG + MEG / 2);   // 8192x64   (0.25M fl)
    __bf16* lnbb  = (__bf16*)(ws + 26 * MEG + 3 * MEG / 4); // 8192x256 (1M fl)
    // window 27.75M..29.75M: libpb early, then gbb
    __bf16* libpb = (__bf16*)(ws + 27 * MEG + 3 * MEG / 4); // 8192x128 (0.5M fl)
    __bf16* gbb   = (__bf16*)(ws + 27 * MEG + 3 * MEG / 4); // 8192x512 (2M fl)
    __bf16* wbufb = (__bf16*)(ws + 29 * MEG + 3 * MEG / 4); // 8192x256 (1M fl)
    __bf16* lbufb = (__bf16*)(ws + 30 * MEG + 3 * MEG / 4); // 8192x256 (1M fl)
    __bf16* fbufb = (__bf16*)(ws + 31 * MEG + 3 * MEG / 4); // 8192x256 (1M fl)
    // weights, converted to bf16 (per-layer ones reused each iteration)
    float* wbase  = ws + 32 * MEG + 3 * MEG / 4;
    __bf16* w2vWb  = (__bf16*)(wbase);            // 256x768  ( 98304 fl)
    __bf16* libWb  = (__bf16*)(wbase + 98304);    // 256x128  ( 16384 fl)
    __bf16* fuseWb = (__bf16*)(wbase + 114688);   // 256x512  ( 65536 fl)
    __bf16* projWb = (__bf16*)(wbase + 180224);   // 256x256  ( 32768 fl)
    __bf16* inWb   = (__bf16*)(wbase + 212992);   // 1024x256 (131072 fl)
    __bf16* xprWb  = (__bf16*)(wbase + 344064);   // 64x512   ( 16384 fl)
    __bf16* dtWb   = (__bf16*)(wbase + 360448);   // 512x64   ( 16384 fl)
    __bf16* outWb  = (__bf16*)(wbase + 376832);   // 256x512  ( 65536 fl)

    const int TOT = M * 512;
    const int EB  = (TOT + 255) / 256;

    // ---- one-time conversions: inputs + shared weights -> bf16 (padded)
    launch_cvt(wav,   M,   768, wavb,  M,   768, stream);
    launch_cvt(lib,   M,    93, libpb, M,   128, stream);
    launch_cvt(w2v_W, 256, 768, w2vWb, 256, 768, stream);
    launch_cvt(lib_W, 256,  93, libWb, 256, 128, stream);
    launch_cvt(fuse_W,256, 512, fuseWb,256, 512, stream);
    launch_cvt(proj_W,256, 256, projWb,256, 256, stream);

    // ---- input projections / fusion
    launch_gemm(wavb,  768, w2vWb, 768, nullptr, w2v_b, nullptr, wbufb, 256,
                M, 256, 768, 0, stream);
    launch_gemm(libpb, 128, libWb, 128, nullptr, lib_b, nullptr, lbufb, 256,
                M, 256, 128, 0, stream);
    launch_gemm(wbufb, 256, fuseWb,       512, nullptr, fuse_b, fbuf, nullptr, 256,
                M, 256, 256, 0, stream);
    launch_gemm(lbufb, 256, fuseWb + 256, 512, fbuf,    nullptr, nullptr, fbufb, 256,
                M, 256, 256, 0, stream);
    launch_gemm(fbufb, 256, projWb, 256, nullptr, proj_b, h, nullptr, 256,
                M, 256, 256, 0, stream);

    // ---- 4 Mamba blocks
    for (int i = 0; i < 4; ++i) {
        // per-layer weights -> bf16 (xproj rows 48->64, dt_W cols 16->64 zero-pad)
        launch_cvt(in_W    + (size_t)i * 1024 * 256, 1024, 256, inWb, 1024, 256, stream);
        launch_cvt(xproj_W + (size_t)i * 48 * 512,   48,  512, xprWb,  64, 512, stream);
        launch_cvt(dt_W    + (size_t)i * 512 * 16,   512,  16, dtWb,  512,  64, stream);
        launch_cvt(out_W   + (size_t)i * 256 * 512,  256, 512, outWb, 256, 512, stream);

        layernorm_kernel<<<M / 8, 256, 0, stream>>>(h, ln_g + i * 256, ln_b + i * 256,
                                                    nullptr, lnbb, M);
        launch_gemm(lnbb, 256, inWb, 256, nullptr, nullptr, xz, nullptr, 1024,
                    M, 1024, 256, 0, stream);
        conv_silu_kernel<<<EB, 256, 0, stream>>>(xz, conv_W + (size_t)i * 512 * 4,
                                                 conv_b + (size_t)i * 512, xc, xcb, TOT);
        // dbl[8192 x 64]: cols 0..15 dt-rank, 16..31 B, 32..47 C, 48..63 zeros
        launch_gemm(xcb, 512, xprWb, 512, nullptr, nullptr, dbl, dblb, 64,
                    M, 64, 512, 0, stream);
        // dt = softplus(dbl[:, :16] @ dt_W^T + dt_b); K padded to 64 (zero weights)
        launch_gemm(dblb, 64, dtWb, 64, nullptr, dt_b + (size_t)i * 512,
                    dtb, nullptr, 512, M, 512, 64, /*softplus*/1, stream);
        scan_kernel<<<8, 512, 0, stream>>>(dtb, xc, dbl, A_log + (size_t)i * 512 * 16, ysb);
        gate_kernel<<<EB, 256, 0, stream>>>(ysb, xc, xz, D_vec + (size_t)i * 512, gbb, TOT);
        launch_gemm(gbb, 512, outWb, 512, h, nullptr, h, nullptr, 256,
                    M, 256, 512, 0, stream);
    }

    // ---- final norm -> output
    layernorm_kernel<<<M / 8, 256, 0, stream>>>(h, fnorm_g, fnorm_b,
                                                (float*)d_out, nullptr, M);
}
